// GNNML3_2181843387147
// MI455X (gfx1250) — matrix operand order repaired and checked by reading, not compiled, not run on hardware
//
#include <hip/hip_runtime.h>
#include <hip/hip_bf16.h>
#include <math.h>

#define N_NODES 50000
#define N_EDGES 1600000
#define NEK     8          // spectral supports / edge feature dim
#define NOUT1   32
#define NOUT2   16
#define XPAD    64         // padded node-feature row (nin <= 48, K padded to mult of 32)
#define HCOLS   256        // NEK * NOUT1

typedef __attribute__((ext_vector_type(16))) _Float16 v16h;
typedef __attribute__((ext_vector_type(8)))  float    v8f;
typedef __attribute__((ext_vector_type(4)))  float    v4f;

// ---------------- utility kernels ----------------

__global__ void prep_x_kernel(const float* __restrict__ x, float* __restrict__ xpad) {
    int idx = blockIdx.x * blockDim.x + threadIdx.x;
    if (idx >= N_NODES * XPAD) return;
    int n = idx >> 6, c = idx & 63;
    xpad[idx] = (c < 32) ? x[n * 32 + c] : 0.f;
}

__global__ void zero_kernel(float* __restrict__ p, int n) {
    int i = blockIdx.x * blockDim.x + threadIdx.x;
    if (i < n) p[i] = 0.f;
}

// convW [K=8][nin][32] (f32) -> pre-swizzled WMMA B fragments (f16):
//   Wh[((nt*2 + kc)*32 + lane)*16 + e] = W2d[kk][col]
//   col = nt*16 + (lane&15)  (output column, = k_support*32 + o)
//   kk  = kc*32 + (e>>3)*16 + (lane>>4)*8 + (e&7)  (input feature row; zero-padded past nin)
// so the GEMM loads one v16h (2x global_load_b128) per B fragment.
__global__ void convw_prep_kernel(const float* __restrict__ convW, _Float16* __restrict__ Wh, int nin) {
    int idx = blockIdx.x * blockDim.x + threadIdx.x;   // 16384 threads
    if (idx >= XPAD * HCOLS) return;
    int e    = idx & 15;
    int lane = (idx >> 4) & 31;
    int kc   = (idx >> 9) & 1;
    int nt   = idx >> 10;
    int col  = nt * 16 + (lane & 15);
    int kk   = kc * 32 + ((e & 8) << 1) + ((lane >> 4) << 3) + (e & 7);
    float v = (kk < nin) ? convW[((col >> 5) * nin + kk) * NOUT1 + (col & 31)] : 0.f;
    Wh[idx] = (_Float16)v;
}

// ---------------- WMMA GEMM: h[N,256] = xpad[N,KP] @ W[KP,256] ----------------
// one block = one 16-row M tile; 8 waves cover the 16 N-tiles (2 each)
__global__ __launch_bounds__(256) void conv_gemm_kernel(const float* __restrict__ xpad,
                                                        const _Float16* __restrict__ Wh,
                                                        float* __restrict__ h, int kloops) {
    int lane  = threadIdx.x & 31;
    int wave  = threadIdx.x >> 5;                 // 0..7
    int m0    = blockIdx.x << 4;                  // 3125 tiles * 16 = 50000 exactly
    int mrow  = m0 + (lane & 15);
    int khalf = lane >> 4;
    int nc    = lane & 15;
    int n0    = wave << 4;
    int n1    = (wave + 8) << 4;

    const v16h* bw = (const v16h*)Wh;             // pre-swizzled fragments, 32B aligned

    v8f c0 = {}; v8f c1 = {};
    for (int kc = 0; kc < kloops; ++kc) {
        // A fragment: per-lane 16 K-values = two contiguous 8-float runs (ISA 7.12.2)
        const float* xp = xpad + (size_t)mrow * XPAD + kc * 32;
        v4f r0 = *(const v4f*)(xp + (khalf << 3));
        v4f r1 = *(const v4f*)(xp + (khalf << 3) + 4);
        v4f r2 = *(const v4f*)(xp + 16 + (khalf << 3));
        v4f r3 = *(const v4f*)(xp + 16 + (khalf << 3) + 4);
        v16h a;
        #pragma unroll
        for (int t = 0; t < 4; ++t) {
            a[t]      = (_Float16)r0[t];
            a[4 + t]  = (_Float16)r1[t];
            a[8 + t]  = (_Float16)r2[t];
            a[12 + t] = (_Float16)r3[t];
        }
        // B fragments: single 32B vector load each
        v16h b0 = bw[(size_t)(wave * 2 + kc) * 32 + lane];
        v16h b1 = bw[(size_t)((wave + 8) * 2 + kc) * 32 + lane];

        c0 = __builtin_amdgcn_wmma_f32_16x16x32_f16(false, a, false, b0, (short)0, c0, false, false);
        c1 = __builtin_amdgcn_wmma_f32_16x16x32_f16(false, a, false, b1, (short)0, c1, false, false);
    }
    // D layout: lane&15 = N col; VGPR r -> M = r + 8*(lane>>4)
    int mbase = m0 + (khalf << 3);
    #pragma unroll
    for (int r = 0; r < 8; ++r) {
        h[(size_t)(mbase + r) * HCOLS + n0 + nc] = c0[r];
        h[(size_t)(mbase + r) * HCOLS + n1 + nc] = c1[r];
    }
}

// ---------------- fused edge MLP + gather + scatter-add (one wave per edge) ----------------
__global__ __launch_bounds__(256) void edge_msg_kernel(const float* __restrict__ ea,
                                                       const int* __restrict__ ei,
                                                       const float* __restrict__ h,
                                                       float* __restrict__ agg,
                                                       const float* __restrict__ e11,
                                                       const float* __restrict__ e12,
                                                       const float* __restrict__ e13,
                                                       const float* __restrict__ e14) {
    __shared__ float s11[NEK * 16], s12[NEK * 16], s13[NEK * 16], s14[32 * NEK];
    for (int i = threadIdx.x; i < NEK * 16; i += blockDim.x) {
        s11[i] = e11[i]; s12[i] = e12[i]; s13[i] = e13[i];
    }
    for (int i = threadIdx.x; i < 32 * NEK; i += blockDim.x) s14[i] = e14[i];
    __syncthreads();

    int edge = (int)((blockIdx.x * blockDim.x + threadIdx.x) >> 5);
    int lane = threadIdx.x & 31;
    if (edge >= N_EDGES) return;

    // wave-uniform broadcast loads of the 8 edge attrs (2x b128)
    const float* eap = ea + (size_t)edge * NEK;
    v4f ea0 = *(const v4f*)(eap);
    v4f ea1 = *(const v4f*)(eap + 4);
    float eav[NEK] = {ea0[0], ea0[1], ea0[2], ea0[3], ea1[0], ea1[1], ea1[2], ea1[3]};

    // hidden u[32]: lanes 0-15 = relu(ea@e11), lanes 16-31 = (ea@e12)*(ea@e13)
    float u;
    if (lane < 16) {
        float t = 0.f;
        #pragma unroll
        for (int k = 0; k < NEK; ++k) t = fmaf(eav[k], s11[k * 16 + lane], t);
        u = fmaxf(t, 0.f);
    } else {
        int j = lane - 16;
        float t2 = 0.f, t3 = 0.f;
        #pragma unroll
        for (int k = 0; k < NEK; ++k) {
            t2 = fmaf(eav[k], s12[k * 16 + j], t2);
            t3 = fmaf(eav[k], s13[k * 16 + j], t3);
        }
        u = t2 * t3;
    }

    // e_k = relu( sum_j u_j * e14[j][k] ): wave-wide xor-tree reduction of 8-vectors
    float ev[NEK];
    #pragma unroll
    for (int k = 0; k < NEK; ++k) ev[k] = u * s14[lane * NEK + k];
    #pragma unroll
    for (int off = 16; off >= 1; off >>= 1) {
        #pragma unroll
        for (int k = 0; k < NEK; ++k) ev[k] += __shfl_xor(ev[k], off, 32);
    }
    #pragma unroll
    for (int k = 0; k < NEK; ++k) ev[k] = fmaxf(ev[k], 0.f);

    int src = ei[edge];
    int dst = ei[N_EDGES + edge];
    const float* hp = h + (size_t)src * HCOLS + lane;   // coalesced 128B per k
    float acc = 0.f;
    #pragma unroll
    for (int k = 0; k < NEK; ++k) acc = fmaf(ev[k], hp[k * 32], acc);
    atomicAdd(&agg[(size_t)dst * 32 + lane], acc);
}

// ---------------- per-node: relu(agg+convB) || tanh-gate, written back into xpad ----------------
__global__ __launch_bounds__(256) void node_update_kernel(float* __restrict__ xpad,
                                                          const float* __restrict__ agg,
                                                          const float* __restrict__ convB,
                                                          const float* __restrict__ f11w,
                                                          const float* __restrict__ f11b,
                                                          const float* __restrict__ f12w,
                                                          const float* __restrict__ f12b, int nin) {
    __shared__ float sW1[48 * 16], sW2[48 * 16], sb1[16], sb2[16], sCB[32];
    for (int i = threadIdx.x; i < nin * 16; i += blockDim.x) { sW1[i] = f11w[i]; sW2[i] = f12w[i]; }
    if (threadIdx.x < 16) { sb1[threadIdx.x] = f11b[threadIdx.x]; sb2[threadIdx.x] = f12b[threadIdx.x]; }
    if (threadIdx.x < 32) sCB[threadIdx.x] = convB[threadIdx.x];
    __syncthreads();

    int n = blockIdx.x * blockDim.x + threadIdx.x;
    if (n >= N_NODES) return;
    float* xp = xpad + (size_t)n * XPAD;

    float xr[48];
    for (int i = 0; i < nin; ++i) xr[i] = xp[i];

    float g1[16], g2[16];
    #pragma unroll
    for (int j = 0; j < 16; ++j) { g1[j] = 0.f; g2[j] = 0.f; }
    for (int i = 0; i < nin; ++i) {
        float xi = xr[i];
        #pragma unroll
        for (int j = 0; j < 16; ++j) {
            g1[j] = fmaf(xi, sW1[i * 16 + j], g1[j]);
            g2[j] = fmaf(xi, sW2[i * 16 + j], g2[j]);
        }
    }

    const float* ap = agg + (size_t)n * 32;
    #pragma unroll
    for (int j = 0; j < 32; ++j) xp[j] = fmaxf(ap[j] + sCB[j], 0.f);
    #pragma unroll
    for (int j = 0; j < 16; ++j)
        xp[32 + j] = tanhf(g1[j] + sb1[j]) * tanhf(g2[j] + sb2[j]);
    #pragma unroll
    for (int j = 48; j < 64; ++j) xp[j] = 0.f;   // keep K-padding clean for next GEMM
}

// ---------------- pooling + head ----------------
__global__ void pool_accum_kernel(const float* __restrict__ xpad, const int* __restrict__ batch,
                                  float* __restrict__ sums, float* __restrict__ cnt) {
    int n = blockIdx.x * blockDim.x + threadIdx.x;
    if (n >= N_NODES) return;
    int g = batch[n];
    const float* xp = xpad + (size_t)n * XPAD;
    #pragma unroll
    for (int c = 0; c < 48; ++c) atomicAdd(&sums[g * 48 + c], xp[c]);
    atomicAdd(&cnt[g], 1.0f);
}

__global__ void head_kernel(const float* __restrict__ sums, const float* __restrict__ cnt,
                            const float* __restrict__ fc1w, const float* __restrict__ fc1b,
                            const float* __restrict__ fc2w, const float* __restrict__ fc2b,
                            float* __restrict__ out) {
    int g = threadIdx.x;
    if (g >= 64) return;
    float inv = 1.0f / fmaxf(cnt[g], 1.0f);
    float p[48];
    #pragma unroll
    for (int i = 0; i < 48; ++i) p[i] = sums[g * 48 + i] * inv;
    float o = fc2b[0];
    #pragma unroll
    for (int j = 0; j < 10; ++j) {
        float hj = fc1b[j];
        for (int i = 0; i < 48; ++i) hj = fmaf(p[i], fc1w[i * 10 + j], hj);
        o = fmaf(fmaxf(hj, 0.f), fc2w[j], o);
    }
    out[g] = o;
}

// ---------------- host orchestration ----------------
extern "C" void kernel_launch(void* const* d_in, const int* in_sizes, int n_in,
                              void* d_out, int out_size, void* d_ws, size_t ws_size,
                              hipStream_t stream) {
    (void)in_sizes; (void)n_in; (void)out_size; (void)ws_size;
    const float* x     = (const float*)d_in[0];
    const int*   ei    = (const int*)d_in[1];
    const float* ea    = (const float*)d_in[2];
    const int*   batch = (const int*)d_in[3];

    // workspace layout (f32 unless noted)
    float* xpad   = (float*)d_ws;                          // N*64      = 3,200,000
    float* hbuf   = xpad + (size_t)N_NODES * XPAD;         // N*256     = 12,800,000
    float* agg    = hbuf + (size_t)N_NODES * HCOLS;        // N*32      = 1,600,000
    float* pooled = agg + (size_t)N_NODES * 32;            // 64*48
    float* cnt    = pooled + 64 * 48;                      // 64
    _Float16* Wh  = (_Float16*)(cnt + 64);                 // 64*256 f16 (32B-aligned offset)

    prep_x_kernel<<<(N_NODES * XPAD + 255) / 256, 256, 0, stream>>>(x, xpad);

    for (int l = 0; l < 3; ++l) {
        int pb  = 4 + 10 * l;
        int nin = (l == 0) ? 32 : 48;
        int kloops = (l == 0) ? 1 : 2;    // K padded to 32 / 64
        const float* e11   = (const float*)d_in[pb + 0];
        const float* e12   = (const float*)d_in[pb + 1];
        const float* e13   = (const float*)d_in[pb + 2];
        const float* e14   = (const float*)d_in[pb + 3];
        const float* convW = (const float*)d_in[pb + 4];
        const float* convB = (const float*)d_in[pb + 5];
        const float* f11w  = (const float*)d_in[pb + 6];
        const float* f11b  = (const float*)d_in[pb + 7];
        const float* f12w  = (const float*)d_in[pb + 8];
        const float* f12b  = (const float*)d_in[pb + 9];

        convw_prep_kernel<<<(XPAD * HCOLS + 255) / 256, 256, 0, stream>>>(convW, Wh, nin);
        conv_gemm_kernel<<<N_NODES / 16, 256, 0, stream>>>(xpad, Wh, hbuf, kloops);
        zero_kernel<<<(N_NODES * 32 + 255) / 256, 256, 0, stream>>>(agg, N_NODES * 32);
        edge_msg_kernel<<<(N_EDGES * 32) / 256, 256, 0, stream>>>(ea, ei, hbuf, agg, e11, e12, e13, e14);
        node_update_kernel<<<(N_NODES + 255) / 256, 256, 0, stream>>>(xpad, agg, convB,
                                                                      f11w, f11b, f12w, f12b, nin);
    }

    zero_kernel<<<(64 * 49 + 255) / 256, 256, 0, stream>>>(pooled, 64 * 49);  // pooled + cnt contiguous
    pool_accum_kernel<<<(N_NODES + 255) / 256, 256, 0, stream>>>(xpad, batch, pooled, cnt);
    head_kernel<<<1, 64, 0, stream>>>(pooled, cnt,
                                      (const float*)d_in[34], (const float*)d_in[35],
                                      (const float*)d_in[36], (const float*)d_in[37],
                                      (float*)d_out);
}
